// MultiHeadAttention_33432025432166
// MI455X (gfx1250) — matrix operand order repaired and checked by reading, not compiled, not run on hardware
//
#include <hip/hip_runtime.h>

// ---------------------------------------------------------------------------
// MultiHeadAttention forward for MI455X (gfx1250, wave32, WMMA).
// B=4, S=2048, D=1024, H=16, DK=64.  Outputs: (out [B,S,D] fp32, attn [B,H,S,S] fp32)
// ---------------------------------------------------------------------------

#define CB 4       // batch
#define CS 2048    // sequence
#define CD 1024    // model dim
#define CH 16      // heads
#define CDK 64     // head dim

typedef __bf16 bf16;
typedef __attribute__((ext_vector_type(16))) __bf16 v16bf;
typedef __attribute__((ext_vector_type(8)))  float  v8f;
typedef __attribute__((ext_vector_type(4)))  int    i32x4;

// Native converts: gfx1250 has hardware bf16<->f32; let clang emit v_cvt.
__device__ __forceinline__ bf16 f32_to_bf16(float f) { return (bf16)f; }

// ---- async global -> LDS (CDNA5 path), guarded by toolchain support -------
#if defined(__has_builtin)
#if __has_builtin(__builtin_amdgcn_global_load_async_to_lds_b128)
#define HAVE_ASYNC_LDS 1
#endif
#endif
#ifndef HAVE_ASYNC_LDS
#define HAVE_ASYNC_LDS 0
#endif

__device__ __forceinline__ void wait_async0() {
#if defined(__has_builtin)
#if __has_builtin(__builtin_amdgcn_s_wait_asynccnt)
  __builtin_amdgcn_s_wait_asynccnt(0);
  return;
#endif
#endif
#if HAVE_ASYNC_LDS
  asm volatile("s_wait_asynccnt 0x0" ::: "memory");
#endif
}

// ---------------------------------------------------------------------------
// WMMA fragment loaders, wave32, 16x16x32 bf16 shapes.
// A (16x32, MxK): lane l<16 -> row l, elems 0..15 = K 0..15
//                 lane l>=16 -> row l-16, elems 0..15 = K 16..31
// B (32x16, KxN): lane l -> column N = l&15, K range = (l<16 ? 0..15 : 16..31)
// C/D (16x16):    VGPR r, lane l -> row = r + (l>=16 ? 8 : 0), col = l&15
// ---------------------------------------------------------------------------
__device__ __forceinline__ v16bf load_a_frag_f32(const float* __restrict__ base,
                                                 int ld, int lane) {
  const int r  = lane & 15;
  const int kb = (lane & 16) ? 16 : 0;
  const float* p = base + r * ld + kb;
  v16bf a;
#pragma unroll
  for (int e = 0; e < 8; ++e) a[e]     = f32_to_bf16(p[e]);
#pragma unroll
  for (int e = 0; e < 8; ++e) a[8 + e] = f32_to_bf16(p[8 + e]);
  return a;
}

__device__ __forceinline__ v16bf load_a_frag_bf16(const bf16* __restrict__ base,
                                                  int ld, int lane) {
  const int r  = lane & 15;
  const int kb = (lane & 16) ? 16 : 0;
  const bf16* p = base + r * ld + kb;
  v16bf a;
#pragma unroll
  for (int e = 0; e < 8; ++e) a[e]     = p[e];
#pragma unroll
  for (int e = 0; e < 8; ++e) a[8 + e] = p[8 + e];
  return a;
}

// B operand taken from a row-major source holding B^T (column n of B = row n
// of source) -> fully contiguous 16-element loads per lane.
__device__ __forceinline__ v16bf load_bt_frag_f32(const float* __restrict__ base,
                                                  int ld, int lane) {
  const int n  = lane & 15;
  const int kb = (lane & 16) ? 16 : 0;
  const float* p = base + n * ld + kb;
  v16bf b;
#pragma unroll
  for (int e = 0; e < 16; ++e) b[e] = f32_to_bf16(p[e]);
  return b;
}

__device__ __forceinline__ v16bf load_bt_frag_bf16(const bf16* __restrict__ base,
                                                   int ld, int lane) {
  const int n  = lane & 15;
  const int kb = (lane & 16) ? 16 : 0;
  const bf16* p = base + n * ld + kb;
  v16bf b;
#pragma unroll
  for (int e = 0; e < 16; ++e) b[e] = p[e];
  return b;
}

// B operand from a genuinely row-major KxN tile (strided) -- used with LDS.
__device__ __forceinline__ v16bf load_b_frag_rowmajor(const bf16* base,
                                                      int ld, int lane) {
  const int n  = lane & 15;
  const int kb = (lane & 16) ? 16 : 0;
  v16bf b;
#pragma unroll
  for (int e = 0; e < 16; ++e) b[e] = base[(kb + e) * ld + n];
  return b;
}

#define WMMA_BF16(A, B, C) \
  __builtin_amdgcn_wmma_f32_16x16x32_bf16(false, (A), false, (B), (short)0, (C), false, false)

// ---------------------------------------------------------------------------
// Kernel 1: Q/K/V projections.  Y = X @ W^T + b, X:[B*S, D] fp32, W:[D, D] fp32.
// Writes bf16 head-major [B,H,S,DK].  blockIdx.y selects q/k/v.
// One wave computes a 16x64 output tile.
// ---------------------------------------------------------------------------
__global__ void __launch_bounds__(256)
qkv_proj_kernel(const float* __restrict__ xq, const float* __restrict__ xk,
                const float* __restrict__ xv,
                const float* __restrict__ wq, const float* __restrict__ bq,
                const float* __restrict__ wk, const float* __restrict__ bk,
                const float* __restrict__ wv, const float* __restrict__ bv,
                bf16* __restrict__ Qh, bf16* __restrict__ Kh, bf16* __restrict__ Vh)
{
  const int lane = threadIdx.x;
  const int wave = threadIdx.y;
  const int which = blockIdx.y;

  const float* X    = (which == 0) ? xq : (which == 1) ? xk : xv;
  const float* W    = (which == 0) ? wq : (which == 1) ? wk : wv;
  const float* bias = (which == 0) ? bq : (which == 1) ? bk : bv;
  bf16*        OUT  = (which == 0) ? Qh : (which == 1) ? Kh : Vh;

  const int tile = blockIdx.x * 8 + wave;   // 8192 wave-tiles: 512 mt x 16 nt64
  const int mt = tile >> 4;                 // / (CD/64)
  const int nt = tile & 15;
  const int m0 = mt * 16;
  const int n0 = nt * 64;

  v8f acc[4] = {};
  for (int k0 = 0; k0 < CD; k0 += 32) {
    const float* abase = X + (size_t)m0 * CD + k0;
    if (k0 + 32 < CD)
      __builtin_prefetch(abase + 32, 0, 1);        // global_prefetch_b8
    v16bf a = load_a_frag_f32(abase, CD, lane);
#pragma unroll
    for (int j = 0; j < 4; ++j) {
      v16bf b = load_bt_frag_f32(W + (size_t)(n0 + j * 16) * CD + k0, CD, lane);
      acc[j] = WMMA_BF16(a, b, acc[j]);
    }
  }

  const int ncol  = lane & 15;
  const int rbase = m0 + ((lane & 16) ? 8 : 0);
#pragma unroll
  for (int j = 0; j < 4; ++j) {
    const int col = n0 + j * 16 + ncol;
    const int h  = col >> 6;        // / CDK
    const int dk = col & 63;
    const float bcol = bias[col];
#pragma unroll
    for (int r = 0; r < 8; ++r) {
      const int row = rbase + r;           // [0, B*S)
      const int b_  = row >> 11;           // / CS
      const int s_  = row & (CS - 1);
      const size_t idx = (((size_t)(b_ * CH + h) * CS) + s_) * CDK + dk;
      OUT[idx] = f32_to_bf16(acc[j][r] + bcol);
    }
  }
}

// ---------------------------------------------------------------------------
// Kernel 2: scores = (Q @ K^T) / sqrt(DK), causal masked, fp32 into attn area.
// One wave computes a 16(q) x 64(k) tile; DK loop = 2 WMMA steps.
// ---------------------------------------------------------------------------
__global__ void __launch_bounds__(256)
scores_kernel(const bf16* __restrict__ Qh, const bf16* __restrict__ Kh,
              float* __restrict__ attn)
{
  const int lane = threadIdx.x;
  const int wave = threadIdx.y;
  const int bh   = blockIdx.y;              // 0..63
  const int tile = blockIdx.x * 8 + wave;   // 4096: 128 qt x 32 kt64
  const int qt = tile >> 5;
  const int kt = tile & 31;
  const int q0  = qt * 16;
  const int k0c = kt * 64;

  const bf16* Q = Qh + (size_t)bh * CS * CDK;
  const bf16* K = Kh + (size_t)bh * CS * CDK;

  v8f acc[4] = {};
#pragma unroll
  for (int kk = 0; kk < CDK; kk += 32) {
    v16bf a = load_a_frag_bf16(Q + (size_t)q0 * CDK + kk, CDK, lane);
#pragma unroll
    for (int j = 0; j < 4; ++j) {
      v16bf b = load_bt_frag_bf16(K + (size_t)(k0c + j * 16) * CDK + kk, CDK, lane);
      acc[j] = WMMA_BF16(a, b, acc[j]);
    }
  }

  const float scale = 0.125f;               // 1/sqrt(64)
  const int ncol  = lane & 15;
  const int rbase = q0 + ((lane & 16) ? 8 : 0);
  float* arow = attn + (size_t)bh * CS * CS;
#pragma unroll
  for (int j = 0; j < 4; ++j) {
    const int col = k0c + j * 16 + ncol;
#pragma unroll
    for (int r = 0; r < 8; ++r) {
      const int row = rbase + r;
      float sv = acc[j][r] * scale;
      if (col > row) sv = -1e9f;            // causal mask
      arow[(size_t)row * CS + col] = sv;
    }
  }
}

// ---------------------------------------------------------------------------
// Kernel 3: row softmax over attn in place. One wave32 per row (2048 elems ->
// 64 per lane, held in registers), cross-lane reduce via shfl_xor.
// ---------------------------------------------------------------------------
__global__ void __launch_bounds__(256)
softmax_kernel(float* __restrict__ attn)
{
  const int lane = threadIdx.x;
  const long long row = (long long)blockIdx.x * 8 + threadIdx.y;  // < B*H*S
  float* p = attn + row * CS;

  float v[CS / 32];
  float m = -3.4e38f;
#pragma unroll
  for (int t = 0; t < CS / 32; ++t) {
    v[t] = p[lane + 32 * t];
    m = fmaxf(m, v[t]);
  }
#pragma unroll
  for (int o = 16; o > 0; o >>= 1) m = fmaxf(m, __shfl_xor(m, o, 32));

  float s = 0.0f;
#pragma unroll
  for (int t = 0; t < CS / 32; ++t) {
    v[t] = __expf(v[t] - m);
    s += v[t];
  }
#pragma unroll
  for (int o = 16; o > 0; o >>= 1) s += __shfl_xor(s, o, 32);

  const float inv = 1.0f / s;
#pragma unroll
  for (int t = 0; t < CS / 32; ++t) p[lane + 32 * t] = v[t] * inv;
}

// ---------------------------------------------------------------------------
// Kernel 4: ctx = attn @ V per head, causal-truncated K loop.  Each wave
// computes 16(q) x 64(full DK) and stages the 32x64 V tile in its own LDS
// slice.  On CDNA5 toolchains with async-LDS builtins the stage uses
// GLOBAL_LOAD_ASYNC_TO_LDS_B128 (bypasses VGPRs, tracked by ASYNCcnt, synced
// with s_wait_asynccnt); otherwise a vectorized VGPR bounce copy.
// Writes ctx bf16 in [B*S, D] layout (head dims re-interleaved).
// ---------------------------------------------------------------------------
__global__ void __launch_bounds__(256)
av_kernel(const float* __restrict__ attn, const bf16* __restrict__ Vh,
          bf16* __restrict__ ctx)
{
  __shared__ bf16 vstage[8][32 * CDK];      // 4 KB per wave, 32 KB total

  const int lane = threadIdx.x;
  const int wave = threadIdx.y;
  const int bh   = blockIdx.y;              // 0..63
  const int qt   = blockIdx.x * 8 + wave;   // 0..127
  const int q0   = qt * 16;
  const int b_ = bh >> 4;                   // / CH
  const int h_ = bh & (CH - 1);

  const float* Arow = attn + (size_t)bh * CS * CS;
  const bf16*  V    = Vh   + (size_t)bh * CS * CDK;
  bf16* lds = vstage[wave];

  v8f acc[4] = {};
  const int kmax = q0 + 16;                 // keys beyond diagonal have attn==0
  for (int k0 = 0; k0 < kmax; k0 += 32) {
    // stage V[k0+lane][0..63] -> LDS (one 128B row per lane)
#if HAVE_ASYNC_LDS
    {
      const bf16* srow = V + (size_t)(k0 + lane) * CDK;
      bf16*       drow = lds + lane * CDK;
#pragma unroll
      for (int t = 0; t < 8; ++t)
        __builtin_amdgcn_global_load_async_to_lds_b128(
            (i32x4*)(srow + t * 8), (i32x4*)(drow + t * 8),
            /*imm offset*/ 0, /*cpol*/ 0);
      wait_async0();                        // ASYNCcnt -> 0 before ds reads
      asm volatile("" ::: "memory");
    }
#else
    {
      const uint4* src = (const uint4*)(V + (size_t)(k0 + lane) * CDK);
      uint4* dst = (uint4*)(lds + lane * CDK);
#pragma unroll
      for (int t = 0; t < 8; ++t) dst[t] = src[t];
    }
#endif
    v16bf a = load_a_frag_f32(Arow + (size_t)q0 * CS + k0, CS, lane);
#pragma unroll
    for (int j = 0; j < 4; ++j) {
      v16bf b = load_b_frag_rowmajor(lds + j * 16, CDK, lane);
      acc[j] = WMMA_BF16(a, b, acc[j]);
    }
  }

  const int ncol  = lane & 15;
  const int rbase = q0 + ((lane & 16) ? 8 : 0);
#pragma unroll
  for (int j = 0; j < 4; ++j) {
    const int col = h_ * CDK + j * 16 + ncol;     // column in [0, D)
#pragma unroll
    for (int r = 0; r < 8; ++r) {
      const int s_ = rbase + r;
      const size_t i = (size_t)(b_ * CS + s_);
      ctx[i * CD + col] = f32_to_bf16(acc[j][r]);
    }
  }
}

// ---------------------------------------------------------------------------
// Kernel 5: out = ctx @ Wo^T + bo  (bf16 A, fp32 B/out).
// ---------------------------------------------------------------------------
__global__ void __launch_bounds__(256)
out_proj_kernel(const bf16* __restrict__ ctx, const float* __restrict__ wo,
                const float* __restrict__ bo, float* __restrict__ out)
{
  const int lane = threadIdx.x;
  const int wave = threadIdx.y;
  const int tile = blockIdx.x * 8 + wave;   // 8192: 512 mt x 16 nt64
  const int mt = tile >> 4;
  const int nt = tile & 15;
  const int m0 = mt * 16;
  const int n0 = nt * 64;

  v8f acc[4] = {};
  for (int k0 = 0; k0 < CD; k0 += 32) {
    v16bf a = load_a_frag_bf16(ctx + (size_t)m0 * CD + k0, CD, lane);
#pragma unroll
    for (int j = 0; j < 4; ++j) {
      v16bf b = load_bt_frag_f32(wo + (size_t)(n0 + j * 16) * CD + k0, CD, lane);
      acc[j] = WMMA_BF16(a, b, acc[j]);
    }
  }

  const int ncol  = lane & 15;
  const int rbase = m0 + ((lane & 16) ? 8 : 0);
#pragma unroll
  for (int j = 0; j < 4; ++j) {
    const int col = n0 + j * 16 + ncol;
    const float bcol = bo[col];
#pragma unroll
    for (int r = 0; r < 8; ++r) {
      const int row = rbase + r;
      out[(size_t)row * CD + col] = acc[j][r] + bcol;
    }
  }
}

// ---------------------------------------------------------------------------
// Host-side launcher
// ---------------------------------------------------------------------------
extern "C" void kernel_launch(void* const* d_in, const int* in_sizes, int n_in,
                              void* d_out, int out_size, void* d_ws, size_t ws_size,
                              hipStream_t stream) {
  (void)in_sizes; (void)n_in; (void)out_size; (void)ws_size;

  const float* q    = (const float*)d_in[0];
  const float* k    = (const float*)d_in[1];
  const float* v    = (const float*)d_in[2];
  /* d_in[3] = mask: causal tril, handled analytically */
  const float* wq_w = (const float*)d_in[4];
  const float* wq_b = (const float*)d_in[5];
  const float* wk_w = (const float*)d_in[6];
  const float* wk_b = (const float*)d_in[7];
  const float* wv_w = (const float*)d_in[8];
  const float* wv_b = (const float*)d_in[9];
  const float* wo_w = (const float*)d_in[10];
  const float* wo_b = (const float*)d_in[11];

  float* out      = (float*)d_out;                          // [B,S,D]
  float* attn_out = out + (size_t)CB * CS * CD;             // [B,H,S,S]

  // workspace carve: Qh | Kh | Vh (bf16 head-major) | ctx (bf16 [B*S, D])
  const size_t nqkv = (size_t)CB * CH * CS * CDK;           // 8,388,608
  bf16* Qh  = (bf16*)d_ws;
  bf16* Kh  = Qh + nqkv;
  bf16* Vh  = Kh + nqkv;
  bf16* ctx = Vh + nqkv;

  const dim3 blk(32, 8);

  // 1) Q/K/V projections: 8192 wave-tiles each, blockIdx.y picks matrix.
  qkv_proj_kernel<<<dim3(1024, 3), blk, 0, stream>>>(
      q, k, v, wq_w, wq_b, wk_w, wk_b, wv_w, wv_b, Qh, Kh, Vh);

  // 2) masked scaled scores -> attn area of d_out (raw, pre-softmax).
  scores_kernel<<<dim3(512, 64), blk, 0, stream>>>(Qh, Kh, attn_out);

  // 3) row softmax in place (B*H*S = 131072 rows, 8 rows/block).
  softmax_kernel<<<dim3((CB * CH * CS) / 8), blk, 0, stream>>>(attn_out);

  // 4) ctx = attn @ V (causal-truncated), bf16 ctx in [B*S, D].
  av_kernel<<<dim3(CS / 16 / 8, CB * CH), blk, 0, stream>>>(attn_out, Vh, ctx);

  // 5) out = ctx @ Wo^T + bo.
  out_proj_kernel<<<dim3(1024, 1), blk, 0, stream>>>(ctx, wo_w, wo_b, out);
}
